// MultiLayerAttention_75041668595846
// MI455X (gfx1250) — matrix operand order repaired and checked by reading, not compiled, not run on hardware
//
#include <hip/hip_runtime.h>
#include <math.h>

typedef __bf16 bf16;
typedef __attribute__((ext_vector_type(4)))  float v4f;
typedef __attribute__((ext_vector_type(8)))  float v8f;
typedef __attribute__((ext_vector_type(4)))  __bf16 v4bf;
typedef __attribute__((ext_vector_type(8)))  __bf16 v8bf;
typedef __attribute__((ext_vector_type(16))) __bf16 v16bf;

#define NHEADS 32
#define NKV 8
#define HDIM 64
#define GQA_GROUP 4
#define MASKV (-65504.0f)

static __device__ __forceinline__ v8f wmma_bf16(v16bf a, v16bf b, v8f c) {
  // D(16x16 f32) = A(16x32 bf16) * B(32x16 bf16) + C
  return __builtin_amdgcn_wmma_f32_16x16x32_bf16(
      false, a, false, b, (short)0, c, false, false);
}

static __device__ __forceinline__ v16bf cat8(v8bf a, v8bf b) {
  return __builtin_shufflevector(a, b, 0, 1, 2, 3, 4, 5, 6, 7,
                                 8, 9, 10, 11, 12, 13, 14, 15);
}

static __device__ __forceinline__ void split1(float x, bf16& h, bf16& l) {
  bf16 hh = (bf16)x;
  h = hh;
  l = (bf16)(x - (float)hh);
}

// ---- CDNA5 async memory->LDS DMA (ASYNCcnt-tracked, no VGPR round trip) ----
static __device__ __forceinline__ void async_ld_b128(void* lds, const void* gsrc) {
  // generic pointer of a __shared__ object: low 32 bits = wave LDS byte offset
  unsigned loff = (unsigned)(size_t)lds;
  asm volatile("global_load_async_to_lds_b128 %0, %1, off"
               :: "v"(loff), "v"(gsrc) : "memory");
}
static __device__ __forceinline__ void wait_async0() {
  asm volatile("s_wait_asynccnt 0x0" ::: "memory");
}

// ---------------------------------------------------------------------------
// fp32 -> (hi, lo) bf16 planes, elementwise (n multiple of 4)
// ---------------------------------------------------------------------------
__global__ __launch_bounds__(256) void split_fp32(const float* __restrict__ x,
                                                  bf16* __restrict__ hi,
                                                  bf16* __restrict__ lo,
                                                  int n4) {
  const int i = blockIdx.x * blockDim.x + threadIdx.x;
  if (i >= n4) return;
  v4f v = *(const v4f*)(x + 4 * (size_t)i);
  v4bf h, l;
#pragma unroll
  for (int j = 0; j < 4; ++j) {
    bf16 hh, ll;
    split1(v[j], hh, ll);
    h[j] = hh;
    l[j] = ll;
  }
  *(v4bf*)(hi + 4 * (size_t)i) = h;
  *(v4bf*)(lo + 4 * (size_t)i) = l;
}

// ---------------------------------------------------------------------------
// V [B*NKV][S][64] -> split bf16, transposed to [B*NKV][64][S]
// ---------------------------------------------------------------------------
__global__ __launch_bounds__(256) void split_transpose_v(const float* __restrict__ V,
                                                         bf16* __restrict__ Th,
                                                         bf16* __restrict__ Tl,
                                                         int S) {
  __shared__ bf16 Hs[64][72];
  __shared__ bf16 Ls[64][72];
  const int tid = threadIdx.x;
  int bid = blockIdx.x;
  const int nsb = S >> 6;
  const int sb = bid % nsb; bid /= nsb;
  const int bk = bid;
  const int s0 = sb * 64;
  {
    const int row = tid >> 2;            // key 0..63
    const int d0 = (tid & 3) * 16;       // dim segment
    const float* src = V + ((size_t)bk * S + s0 + row) * HDIM + d0;
#pragma unroll
    for (int u = 0; u < 4; ++u) {
      v4f v = *(const v4f*)(src + 4 * u);
#pragma unroll
      for (int j = 0; j < 4; ++j) {
        bf16 h, l;
        split1(v[j], h, l);
        Hs[d0 + 4 * u + j][row] = h;
        Ls[d0 + 4 * u + j][row] = l;
      }
    }
  }
  __syncthreads();
  {
    const int d = tid >> 2;
    const int k0 = (tid & 3) * 16;
    bf16* dh = Th + ((size_t)bk * HDIM + d) * S + s0 + k0;
    bf16* dl = Tl + ((size_t)bk * HDIM + d) * S + s0 + k0;
    *(v8bf*)(dh)     = *(const v8bf*)&Hs[d][k0];
    *(v8bf*)(dh + 8) = *(const v8bf*)&Hs[d][k0 + 8];
    *(v8bf*)(dl)     = *(const v8bf*)&Ls[d][k0];
    *(v8bf*)(dl + 8) = *(const v8bf*)&Ls[d][k0 + 8];
  }
}

// ---------------------------------------------------------------------------
// Split-bf16 GEMM: C = (Ah+Al) @ (Wh+Wl)^T + bias, async double-buffered LDS.
// Block 128x128, K chunk 32; wave owns 32x64 -> 2x4 acc tiles, 3 WMMAs each.
// SPLIT_OUT: write (hi,lo) bf16 planes scaled by oscale; else f32.
// ---------------------------------------------------------------------------
template <bool SPLIT_OUT>
__global__ __launch_bounds__(256) void gemm_bf16s(
    const bf16* __restrict__ Ah, const bf16* __restrict__ Al,
    const bf16* __restrict__ Wh, const bf16* __restrict__ Wl,
    const float* __restrict__ bias,
    float* __restrict__ Cf, bf16* __restrict__ Ch, bf16* __restrict__ Cl,
    float oscale, int M, int N, int K) {
  __shared__ bf16 Ahs[2][128][40];
  __shared__ bf16 Als[2][128][40];
  __shared__ bf16 Bhs[2][128][40];
  __shared__ bf16 Bls[2][128][40];

  const int tid  = threadIdx.x;
  const int lane = tid & 31;
  const int wid  = tid >> 5;
  const int ln   = lane & 15;
  const int lg   = lane >> 4;
  const int wm   = wid & 3;
  const int wn   = wid >> 2;

  const int m0 = blockIdx.x * 128;
  const int n0 = blockIdx.y * 128;

  v8f acc[2][4];
#pragma unroll
  for (int mi = 0; mi < 2; ++mi)
#pragma unroll
    for (int ni = 0; ni < 4; ++ni)
#pragma unroll
      for (int r = 0; r < 8; ++r) acc[mi][ni][r] = 0.0f;

  const int lrow = tid >> 1;
  const int lseg = (tid & 1) * 16;
  const bf16* Agh = Ah + (size_t)(m0 + lrow) * K + lseg;
  const bf16* Agl = Al + (size_t)(m0 + lrow) * K + lseg;
  const bf16* Wgh = Wh + (size_t)(n0 + lrow) * K + lseg;
  const bf16* Wgl = Wl + (size_t)(n0 + lrow) * K + lseg;

  auto issue = [&](int kk, int bb) {
    async_ld_b128(&Ahs[bb][lrow][lseg],     Agh + kk);
    async_ld_b128(&Ahs[bb][lrow][lseg + 8], Agh + kk + 8);
    async_ld_b128(&Als[bb][lrow][lseg],     Agl + kk);
    async_ld_b128(&Als[bb][lrow][lseg + 8], Agl + kk + 8);
    async_ld_b128(&Bhs[bb][lrow][lseg],     Wgh + kk);
    async_ld_b128(&Bhs[bb][lrow][lseg + 8], Wgh + kk + 8);
    async_ld_b128(&Bls[bb][lrow][lseg],     Wgl + kk);
    async_ld_b128(&Bls[bb][lrow][lseg + 8], Wgl + kk + 8);
  };

  issue(0, 0);
  int buf = 0;
  for (int kk = 0; kk < K; kk += 32) {
    wait_async0();        // my chunk-kk DMA has landed in LDS
    __syncthreads();      // everyone's landed; everyone done with buf^1
    if (kk + 32 < K) {
      issue(kk + 32, buf ^ 1);                   // DMA overlaps compute below
      __builtin_prefetch(Agh + kk + 96, 0, 1);   // prime L2 two chunks past the DMA chunk
      __builtin_prefetch(Wgh + kk + 96, 0, 1);
    }

    // fragments: A lanes0-15 K=0..15, lanes16-31 K=16..31;
    //            B lanes0-15 K=0..15, lanes16-31 K=16..31
    v16bf afh[2], afl[2], bfh[4], bfl[4];
#pragma unroll
    for (int mi = 0; mi < 2; ++mi) {
      const int row = 32 * wm + 16 * mi + ln;
      afh[mi] = cat8(*(const v8bf*)&Ahs[buf][row][16 * lg],
                     *(const v8bf*)&Ahs[buf][row][16 * lg + 8]);
      afl[mi] = cat8(*(const v8bf*)&Als[buf][row][16 * lg],
                     *(const v8bf*)&Als[buf][row][16 * lg + 8]);
    }
#pragma unroll
    for (int ni = 0; ni < 4; ++ni) {
      const int row = 64 * wn + 16 * ni + ln;
      bfh[ni] = cat8(*(const v8bf*)&Bhs[buf][row][16 * lg],
                     *(const v8bf*)&Bhs[buf][row][16 * lg + 8]);
      bfl[ni] = cat8(*(const v8bf*)&Bls[buf][row][16 * lg],
                     *(const v8bf*)&Bls[buf][row][16 * lg + 8]);
    }
#pragma unroll
    for (int mi = 0; mi < 2; ++mi)
#pragma unroll
      for (int ni = 0; ni < 4; ++ni) {
        acc[mi][ni] = wmma_bf16(afh[mi], bfh[ni], acc[mi][ni]);
        acc[mi][ni] = wmma_bf16(afh[mi], bfl[ni], acc[mi][ni]);
        acc[mi][ni] = wmma_bf16(afl[mi], bfh[ni], acc[mi][ni]);
      }
    buf ^= 1;
  }

#pragma unroll
  for (int ni = 0; ni < 4; ++ni) {
    const int col = n0 + 64 * wn + 16 * ni + ln;
    const float bv = bias[col];
#pragma unroll
    for (int mi = 0; mi < 2; ++mi) {
      const int rowb = m0 + 32 * wm + 16 * mi + 8 * lg;
#pragma unroll
      for (int r = 0; r < 8; ++r) {
        const size_t idx = (size_t)(rowb + r) * N + col;
        const float y = acc[mi][ni][r] + bv;
        if constexpr (SPLIT_OUT) {
          bf16 h, l;
          split1(y * oscale, h, l);
          Ch[idx] = h;
          Cl[idx] = l;
        } else {
          Cf[idx] = y;
        }
      }
    }
  }
}

// ---------------------------------------------------------------------------
// Flash attention, causal, GQA; split-bf16 WMMA; async double-buffered K/V.
// ---------------------------------------------------------------------------
__global__ __launch_bounds__(256) void attn_fwd(
    const bf16* __restrict__ Qh, const bf16* __restrict__ Ql,
    const bf16* __restrict__ Kh, const bf16* __restrict__ Kl,
    const bf16* __restrict__ Vth, const bf16* __restrict__ Vtl,
    bf16* __restrict__ Oh, bf16* __restrict__ Ol,
    int B, int S) {
  __shared__ bf16 Khs[2][32][72];     // [key][dim]
  __shared__ bf16 Kls[2][32][72];
  __shared__ bf16 Vhs[2][64][40];     // [dim][key] (pre-transposed in memory)
  __shared__ bf16 Vls[2][64][40];
  __shared__ float Ps[8][16][36];     // per-wave P transpose scratch

  const int H = NHEADS * HDIM;
  const int tid  = threadIdx.x;
  const int lane = tid & 31;
  const int wid  = tid >> 5;
  const int ln   = lane & 15;
  const int lg   = lane >> 4;

  int bid = blockIdx.x;
  const int nqb = S >> 7;
  const int qb = bid % nqb; bid /= nqb;
  const int h  = bid % NHEADS; bid /= NHEADS;
  const int b  = bid;
  const int kvh = h / GQA_GROUP;
  const int q0  = qb * 128;
  const int q0w = q0 + wid * 16;

  // Q A-fragments (pre-scaled by 1/sqrt(64) at projection epilogue)
  v16bf qhf[2], qlf[2];
  {
    const bf16* qph = Qh + ((size_t)b * S + q0w + ln) * H + h * HDIM;
    const bf16* qpl = Ql + ((size_t)b * S + q0w + ln) * H + h * HDIM;
#pragma unroll
    for (int c = 0; c < 2; ++c) {
      const int base = c * 32 + 16 * lg;
      qhf[c] = cat8(*(const v8bf*)(qph + base), *(const v8bf*)(qph + base + 8));
      qlf[c] = cat8(*(const v8bf*)(qpl + base), *(const v8bf*)(qpl + base + 8));
    }
  }

  v8f o[4];
  float Mr[8], Lr[8];
#pragma unroll
  for (int dt = 0; dt < 4; ++dt)
#pragma unroll
    for (int r = 0; r < 8; ++r) o[dt][r] = 0.0f;
#pragma unroll
  for (int r = 0; r < 8; ++r) { Mr[r] = -3.0e38f; Lr[r] = 0.0f; }

  const size_t bkoff = ((size_t)b * NKV + kvh);
  const bf16* kbh = Kh + bkoff * (size_t)S * HDIM;
  const bf16* kbl = Kl + bkoff * (size_t)S * HDIM;
  const bf16* vbh = Vth + bkoff * (size_t)HDIM * S;
  const bf16* vbl = Vtl + bkoff * (size_t)HDIM * S;

  const int krow = tid >> 2;            // tid<128: key row 0..31
  const int kseg = (tid & 3) * 16;
  const int vd   = (tid - 128) >> 1;    // tid>=128: dim 0..63
  const int vseg = ((tid - 128) & 1) * 16;

  auto issue_tile = [&](int kb, int bb) {
    if (tid < 128) {
      const bf16* sh = kbh + (size_t)(kb + krow) * HDIM + kseg;
      const bf16* sl = kbl + (size_t)(kb + krow) * HDIM + kseg;
      async_ld_b128(&Khs[bb][krow][kseg],     sh);
      async_ld_b128(&Khs[bb][krow][kseg + 8], sh + 8);
      async_ld_b128(&Kls[bb][krow][kseg],     sl);
      async_ld_b128(&Kls[bb][krow][kseg + 8], sl + 8);
    } else {
      const bf16* sh = vbh + (size_t)vd * S + kb + vseg;
      const bf16* sl = vbl + (size_t)vd * S + kb + vseg;
      async_ld_b128(&Vhs[bb][vd][vseg],     sh);
      async_ld_b128(&Vhs[bb][vd][vseg + 8], sh + 8);
      async_ld_b128(&Vls[bb][vd][vseg],     sl);
      async_ld_b128(&Vls[bb][vd][vseg + 8], sl + 8);
    }
  };

  const int kend = q0 + 128;
  issue_tile(0, 0);
  int buf = 0;
  for (int kb = 0; kb < kend; kb += 32) {
    wait_async0();
    __syncthreads();
    if (kb + 32 < kend) issue_tile(kb + 32, buf ^ 1);

    if (kb <= q0w + 15) {   // wave-uniform causal skip (EXEC all-ones inside)
      // ---- scores S(16x32) = Q(16x64) @ Ktile^T ----
      v8f s0, s1;
#pragma unroll
      for (int r = 0; r < 8; ++r) { s0[r] = 0.0f; s1[r] = 0.0f; }
#pragma unroll
      for (int c = 0; c < 2; ++c) {
        const int kc = c * 32 + 16 * lg;
        v16bf b0h = cat8(*(const v8bf*)&Khs[buf][ln][kc],
                         *(const v8bf*)&Khs[buf][ln][kc + 8]);
        v16bf b0l = cat8(*(const v8bf*)&Kls[buf][ln][kc],
                         *(const v8bf*)&Kls[buf][ln][kc + 8]);
        v16bf b1h = cat8(*(const v8bf*)&Khs[buf][16 + ln][kc],
                         *(const v8bf*)&Khs[buf][16 + ln][kc + 8]);
        v16bf b1l = cat8(*(const v8bf*)&Kls[buf][16 + ln][kc],
                         *(const v8bf*)&Kls[buf][16 + ln][kc + 8]);
        s0 = wmma_bf16(qhf[c], b0h, s0);
        s0 = wmma_bf16(qhf[c], b0l, s0);
        s0 = wmma_bf16(qlf[c], b0h, s0);
        s1 = wmma_bf16(qhf[c], b1h, s1);
        s1 = wmma_bf16(qhf[c], b1l, s1);
        s1 = wmma_bf16(qlf[c], b1h, s1);
      }

      // ---- causal mask + online softmax ----
      float mnew[8], alpha[8];
#pragma unroll
      for (int r = 0; r < 8; ++r) {
        const int qm = q0w + 8 * lg + r;
        float a = s0[r], c = s1[r];
        if (qm < kb + ln)      a = MASKV;
        if (qm < kb + 16 + ln) c = MASKV;
        s0[r] = a; s1[r] = c;
        float m = fmaxf(a, c);
        m = fmaxf(m, __shfl_xor(m, 1, 32));
        m = fmaxf(m, __shfl_xor(m, 2, 32));
        m = fmaxf(m, __shfl_xor(m, 4, 32));
        m = fmaxf(m, __shfl_xor(m, 8, 32));
        mnew[r] = fmaxf(Mr[r], m);
      }
#pragma unroll
      for (int r = 0; r < 8; ++r) {
        alpha[r] = __expf(Mr[r] - mnew[r]);
        float p0 = __expf(s0[r] - mnew[r]);
        float p1 = __expf(s1[r] - mnew[r]);
        s0[r] = p0; s1[r] = p1;
        float rs = p0 + p1;
        rs += __shfl_xor(rs, 1, 32);
        rs += __shfl_xor(rs, 2, 32);
        rs += __shfl_xor(rs, 4, 32);
        rs += __shfl_xor(rs, 8, 32);
        Lr[r] = Lr[r] * alpha[r] + rs;
        Mr[r] = mnew[r];
      }
#pragma unroll
      for (int dt = 0; dt < 4; ++dt)
#pragma unroll
        for (int r = 0; r < 8; ++r) o[dt][r] *= alpha[r];

      // ---- transpose P through wave-private LDS, split to bf16 A-frags ----
#pragma unroll
      for (int r = 0; r < 8; ++r) {
        Ps[wid][8 * lg + r][ln]      = s0[r];
        Ps[wid][8 * lg + r][16 + ln] = s1[r];
      }
      v16bf ph, pl;
      {
        v4f a0 = *(const v4f*)&Ps[wid][ln][16 * lg];
        v4f a1 = *(const v4f*)&Ps[wid][ln][16 * lg + 4];
        v4f a2 = *(const v4f*)&Ps[wid][ln][16 * lg + 8];
        v4f a3 = *(const v4f*)&Ps[wid][ln][16 * lg + 12];
#pragma unroll
        for (int j = 0; j < 4; ++j) {
          bf16 hh, ll;
          split1(a0[j], hh, ll); ph[j] = hh;      pl[j] = ll;
          split1(a1[j], hh, ll); ph[4 + j] = hh;  pl[4 + j] = ll;
          split1(a2[j], hh, ll); ph[8 + j] = hh;  pl[8 + j] = ll;
          split1(a3[j], hh, ll); ph[12 + j] = hh; pl[12 + j] = ll;
        }
      }

      // ---- O(16x64) += P(16x32) @ Vtile(32x64) ----
#pragma unroll
      for (int dt = 0; dt < 4; ++dt) {
        const int row = dt * 16 + ln;
        v16bf vh = cat8(*(const v8bf*)&Vhs[buf][row][16 * lg],
                        *(const v8bf*)&Vhs[buf][row][16 * lg + 8]);
        v16bf vl = cat8(*(const v8bf*)&Vls[buf][row][16 * lg],
                        *(const v8bf*)&Vls[buf][row][16 * lg + 8]);
        o[dt] = wmma_bf16(ph, vh, o[dt]);
        o[dt] = wmma_bf16(ph, vl, o[dt]);
        o[dt] = wmma_bf16(pl, vh, o[dt]);
      }
    }
    buf ^= 1;
  }

  // ---- normalize, split, store [B,S,H] ----
#pragma unroll
  for (int r = 0; r < 8; ++r) {
    const float inv = 1.0f / Lr[r];
    const size_t rowoff = ((size_t)b * S + q0w + 8 * lg + r) * H + h * HDIM;
#pragma unroll
    for (int dt = 0; dt < 4; ++dt) {
      bf16 hh, ll;
      split1(o[dt][r] * inv, hh, ll);
      Oh[rowoff + dt * 16 + ln] = hh;
      Ol[rowoff + dt * 16 + ln] = ll;
    }
  }
}

// ---------------------------------------------------------------------------
extern "C" void kernel_launch(void* const* d_in, const int* in_sizes, int n_in,
                              void* d_out, int out_size, void* d_ws, size_t ws_size,
                              hipStream_t stream) {
  (void)in_sizes; (void)n_in; (void)out_size; (void)ws_size;
  const float* hidden = (const float*)d_in[0];
  const float* key    = (const float*)d_in[1];
  const float* value  = (const float*)d_in[2];
  const float* Wq     = (const float*)d_in[3];
  const float* bq     = (const float*)d_in[4];
  const float* Wc     = (const float*)d_in[5];
  const float* bc     = (const float*)d_in[6];
  float* out = (float*)d_out;

  const int B = 2, S = 2048, H = 2048;
  const int M = B * S;
  const size_t MH = (size_t)M * H;
  const size_t HH = (size_t)H * H;
  const size_t KV = (size_t)B * NKV * S * HDIM;

  bf16* p = (bf16*)d_ws;
  bf16* hidH = p; p += MH;  bf16* hidL = p; p += MH;
  bf16* WqH  = p; p += HH;  bf16* WqL  = p; p += HH;
  bf16* WcH  = p; p += HH;  bf16* WcL  = p; p += HH;
  bf16* KH   = p; p += KV;  bf16* KL   = p; p += KV;
  bf16* VtH  = p; p += KV;  bf16* VtL  = p; p += KV;
  bf16* qH   = p; p += MH;  bf16* qL   = p; p += MH;
  bf16* aH   = p; p += MH;  bf16* aL   = p; p += MH;   // ~151 MB total

  split_fp32<<<(int)(MH / 4 / 256), 256, 0, stream>>>(hidden, hidH, hidL, (int)(MH / 4));
  split_fp32<<<(int)(HH / 4 / 256), 256, 0, stream>>>(Wq, WqH, WqL, (int)(HH / 4));
  split_fp32<<<(int)(HH / 4 / 256), 256, 0, stream>>>(Wc, WcH, WcL, (int)(HH / 4));
  split_fp32<<<(int)(KV / 4 / 256), 256, 0, stream>>>(key, KH, KL, (int)(KV / 4));
  split_transpose_v<<<B * NKV * (S / 64), 256, 0, stream>>>(value, VtH, VtL, S);

  dim3 gg(M / 128, H / 128);
  // q projection -> split bf16 with 1/sqrt(64) folded in (exact pow2 scale)
  gemm_bf16s<true><<<gg, 256, 0, stream>>>(hidH, hidL, WqH, WqL, bq,
                                           nullptr, qH, qL, 0.125f, M, H, H);
  attn_fwd<<<B * NHEADS * (S / 128), 256, 0, stream>>>(qH, qL, KH, KL, VtH, VtL,
                                                       aH, aL, B, S);
  // c projection -> f32 output
  gemm_bf16s<false><<<gg, 256, 0, stream>>>(aH, aL, WcH, WcL, bc,
                                            out, nullptr, nullptr, 1.0f, M, H, H);
}
